// YOLOv2_80539226735392
// MI455X (gfx1250) — matrix operand order repaired and checked by reading, not compiled, not run on hardware
//
#include <hip/hip_runtime.h>
#include <math.h>

#define CIN   512
#define HW    1024
#define SW    32
#define NBOX  5120
#define NCLS  80
#define KA    5
#define ACT   (512 * 1024)
#define WELEM (512 * 9 * 512)

typedef __attribute__((ext_vector_type(16))) __bf16 v16bf;
typedef __attribute__((ext_vector_type(8)))  float  v8f;

struct __align__(16) U4 { unsigned int a, b, c, d; };
union Frag { U4 q[2]; v16bf v; };

__device__ __forceinline__ unsigned short f2bf(float f) {
  unsigned int u = __builtin_bit_cast(unsigned int, f);
  u += 0x7FFFu + ((u >> 16) & 1u);          // round-to-nearest-even
  return (unsigned short)(u >> 16);
}

__device__ __forceinline__ float sigmoidf(float x) { return 1.0f / (1.0f + expf(-x)); }

__constant__ float c_anchors[KA][2] = {
  {32.0f, 32.0f}, {64.0f, 128.0f}, {128.0f, 64.0f}, {192.0f, 288.0f}, {320.0f, 320.0f}
};

// ---------------------------------------------------------------------------
// Prep: feat f32 [512][1024] -> bf16 transposed [1024][512]; zero the pad col.
// ---------------------------------------------------------------------------
__global__ void featprep(const float* __restrict__ feat,
                         unsigned short* __restrict__ featbf,
                         unsigned short* __restrict__ zerocol)
{
  const int idx = blockIdx.x * blockDim.x + threadIdx.x;
  if (idx >= ACT) return;
  const int n = idx >> 9, c = idx & 511;
  featbf[idx] = f2bf(feat[(size_t)c * HW + n]);
  if (idx < CIN) zerocol[idx] = 0;
}

// ---------------------------------------------------------------------------
// Prep: conv weights f32 [co][ci][3][3] -> bf16 [co][kp][ci]
// (A-fragment rows become K-contiguous => 16B b128 fragment loads)
// ---------------------------------------------------------------------------
__global__ void wprep(const float* __restrict__ w, unsigned short* __restrict__ wbf)
{
  const int idx = blockIdx.x * blockDim.x + threadIdx.x;
  if (idx >= WELEM) return;
  const int co = idx / 4608;
  const int r  = idx - co * 4608;
  const int kp = r >> 9, ci = r & 511;
  wbf[idx] = f2bf(w[((size_t)co * CIN + ci) * 9 + kp]);
}

// ---------------------------------------------------------------------------
// 3x3 conv partial GEMM via direct-fragment WMMA (no LDS staging).
// blockIdx.z = kernel-row group g (kp = 3g..3g+2), writes partial[g].
// Block = 256 thr (8 waves), tile 128x128, wave tile 64x32 (4x2 accums).
// ---------------------------------------------------------------------------
__global__ __launch_bounds__(256) void conv3x3_frag(
    const unsigned short* __restrict__ xbf,     // [1024 hw][512 ci] bf16 (act^T)
    const unsigned short* __restrict__ wbf,     // [512 co][9 kp][512 ci] bf16
    const unsigned short* __restrict__ zerocol, // [512] bf16 zeros
    float* __restrict__ partial)                // [3][512][1024] f32
{
  const int tid  = threadIdx.x;
  const int lane = tid & 31;
  const int wv   = tid >> 5;
  const int wm   = wv >> 2;   // 0..1
  const int wn   = wv & 3;    // 0..3
  const int mblk = blockIdx.y * 128;
  const int nblk = blockIdx.x * 128;
  const int g    = blockIdx.z;            // 0..2
  const int hi   = lane >> 4;
  const int lr   = lane & 15;

  v8f acc[4][2];
#pragma unroll
  for (int i = 0; i < 4; ++i)
#pragma unroll
    for (int j = 0; j < 2; ++j)
#pragma unroll
      for (int e = 0; e < 8; ++e) acc[i][j][e] = 0.0f;

  for (int kq = 0; kq < 3; ++kq) {
    const int kp = g * 3 + kq;
    const int dy = kp / 3 - 1;
    const int dx = kp % 3 - 1;

    // A row base (per-lane): row r, K-chunk start hi*16 (contiguous K, matches B)
    const unsigned short* pA[4];
#pragma unroll
    for (int i = 0; i < 4; ++i) {
      const int r = mblk + wm * 64 + i * 16 + lr;
      pA[i] = wbf + ((size_t)r * 9 + kp) * CIN + hi * 16;
    }
    // B column base (per-lane): shifted column, K-chunk start hi*16
    const unsigned short* pB[2];
#pragma unroll
    for (int j = 0; j < 2; ++j) {
      const int c  = nblk + wn * 32 + j * 16 + lr;
      const int yy = (c >> 5) + dy;
      const int xx = (c & 31) + dx;
      const bool ok = ((unsigned)yy < 32u) && ((unsigned)xx < 32u);
      pB[j] = ok ? (xbf + (size_t)(yy * SW + xx) * CIN + hi * 16)
                 : (zerocol + hi * 16);
    }
    __builtin_prefetch(pA[0], 0, 0);   // global_prefetch_b8 warms weight stream

    for (int ks = 0; ks < 16; ++ks) {
      const int ko = ks * 32;          // halfs
      Frag af[4], bfr[2];
#pragma unroll
      for (int i = 0; i < 4; ++i) {
        af[i].q[0] = *(const U4*)(pA[i] + ko);        // K hi*16..+7
        af[i].q[1] = *(const U4*)(pA[i] + ko + 8);    // K hi*16+8..+15
      }
#pragma unroll
      for (int j = 0; j < 2; ++j) {
        bfr[j].q[0] = *(const U4*)(pB[j] + ko);       // K hi*16..+7
        bfr[j].q[1] = *(const U4*)(pB[j] + ko + 8);   // K hi*16+8..+15
      }
#pragma unroll
      for (int i = 0; i < 4; ++i)
#pragma unroll
        for (int j = 0; j < 2; ++j)
          acc[i][j] = __builtin_amdgcn_wmma_f32_16x16x32_bf16(
              false, af[i].v, false, bfr[j].v, (short)0, acc[i][j], false, false);
    }
  }

  // C layout: (M = hi*8 + vgpr, N = lane&15)
  float* pout = partial + (size_t)g * ACT;
#pragma unroll
  for (int i = 0; i < 4; ++i)
#pragma unroll
    for (int j = 0; j < 2; ++j)
#pragma unroll
      for (int e = 0; e < 8; ++e) {
        const int m = mblk + wm * 64 + i * 16 + hi * 8 + e;
        const int n = nblk + wn * 32 + j * 16 + lr;
        pout[(size_t)m * HW + n] = acc[i][j][e];
      }
}

// ---------------------------------------------------------------------------
// Combine 3 split-K partials + bias + leaky_relu(0.1); emit f32 [c][n] for
// heads and bf16 transposed [n][c] for the next conv's B fragments.
// ---------------------------------------------------------------------------
__global__ void combine_act(const float* __restrict__ p, const float* __restrict__ bias,
                            float* __restrict__ outf, unsigned short* __restrict__ outbfT)
{
  const int idx = blockIdx.x * blockDim.x + threadIdx.x;
  if (idx >= ACT) return;
  const int m = idx >> 10, n = idx & 1023;
  float v = p[idx] + p[ACT + idx] + p[2 * ACT + idx] + bias[m];
  v = (v >= 0.0f) ? v : 0.1f * v;
  outf[idx] = v;
  outbfT[(size_t)n * CIN + m] = f2bf(v);
}

// ---------------------------------------------------------------------------
// 1x1 conv heads (tiny): one thread per (co, n).
// ---------------------------------------------------------------------------
__global__ void conv1x1(const float* __restrict__ act, const float* __restrict__ w,
                        const float* __restrict__ b, float* __restrict__ out, int Co)
{
  const int idx = blockIdx.x * blockDim.x + threadIdx.x;
  if (idx >= Co * HW) return;
  const int co = idx / HW, n = idx % HW;
  const float* wr = w + (size_t)co * CIN;
  float s = b[co];
  for (int c = 0; c < CIN; ++c) s += wr[c] * act[(size_t)c * HW + n];
  out[(size_t)co * HW + n] = s;
}

// ---------------------------------------------------------------------------
// Decode boxes + scores; prep NMS state. d_out row i: [x1 y1 x2 y2, 80 scores]
// ---------------------------------------------------------------------------
__global__ void decode_kernel(const float* __restrict__ obj_raw,
                              const float* __restrict__ cls_raw,
                              const float* __restrict__ reg_raw,
                              float* __restrict__ out,
                              float* __restrict__ sc, int* __restrict__ clsi,
                              int* __restrict__ validf,
                              int* __restrict__ suppressed, int* __restrict__ visited)
{
  const int i = blockIdx.x * blockDim.x + threadIdx.x;
  if (i >= NBOX) return;
  const int n = i / KA, a = i % KA;

  const float t0 = reg_raw[(a * 4 + 0) * HW + n];
  const float t1 = reg_raw[(a * 4 + 1) * HW + n];
  const float t2 = reg_raw[(a * 4 + 2) * HW + n];
  const float t3 = reg_raw[(a * 4 + 3) * HW + n];
  const float gx = (float)(n & 31), gy = (float)(n >> 5);
  const float cx = (sigmoidf(t0) + gx) * 32.0f;
  const float cy = (sigmoidf(t1) + gy) * 32.0f;
  const float bw = expf(t2) * c_anchors[a][0];
  const float bh = expf(t3) * c_anchors[a][1];
  float x1 = (cx - 0.5f * bw) * (1.0f / 1024.0f);
  float y1 = (cy - 0.5f * bh) * (1.0f / 1024.0f);
  float x2 = (cx + 0.5f * bw) * (1.0f / 1024.0f);
  float y2 = (cy + 0.5f * bh) * (1.0f / 1024.0f);
  x1 = fminf(fmaxf(x1, 0.0f), 1.0f); y1 = fminf(fmaxf(y1, 0.0f), 1.0f);
  x2 = fminf(fmaxf(x2, 0.0f), 1.0f); y2 = fminf(fmaxf(y2, 0.0f), 1.0f);

  float* row = out + (size_t)i * 84;
  row[0] = x1; row[1] = y1; row[2] = x2; row[3] = y2;

  const float o = sigmoidf(obj_raw[a * HW + n]);
  float mx = -1e30f;
  for (int c = 0; c < NCLS; ++c) mx = fmaxf(mx, cls_raw[(a * NCLS + c) * HW + n]);
  float sum = 0.0f;
  for (int c = 0; c < NCLS; ++c) sum += expf(cls_raw[(a * NCLS + c) * HW + n] - mx);
  const float inv = 1.0f / sum;
  float best = -1.0f; int barg = 0;
  for (int c = 0; c < NCLS; ++c) {
    const float s = o * expf(cls_raw[(a * NCLS + c) * HW + n] - mx) * inv;
    row[4 + c] = s;
    if (s > best) { best = s; barg = c; }
  }
  sc[i] = best;
  clsi[i] = barg;
  validf[i] = (best >= 0.001f) ? 1 : 0;
  suppressed[i] = 0;
  visited[i] = 0;
}

// ---------------------------------------------------------------------------
// Greedy NMS, single 1024-thread block; iterative stable argmax == the
// reference's stable descending-score serial loop.
// ---------------------------------------------------------------------------
__global__ __launch_bounds__(1024) void nms_kernel(
    const float* __restrict__ out, const float* __restrict__ sc,
    const int* __restrict__ clsi, const int* __restrict__ validf,
    int* __restrict__ suppressed, int* __restrict__ visited,
    float* __restrict__ keep)
{
  __shared__ float sv[1024];
  __shared__ int   si[1024];
  __shared__ int   s_kept;
  const int tid = threadIdx.x;

  for (int it = 0; it < NBOX; ++it) {
    float best = -1e30f; int bi = NBOX;
    for (int j = tid; j < NBOX; j += 1024) {
      if (!visited[j]) {
        const float s = sc[j];
        if (s > best || (s == best && j < bi)) { best = s; bi = j; }
      }
    }
    sv[tid] = best; si[tid] = bi;
    __syncthreads();
    for (int off = 512; off > 0; off >>= 1) {
      if (tid < off) {
        if (sv[tid + off] > sv[tid] ||
            (sv[tid + off] == sv[tid] && si[tid + off] < si[tid])) {
          sv[tid] = sv[tid + off]; si[tid] = si[tid + off];
        }
      }
      __syncthreads();
    }
    const int idx = si[0];
    if (tid == 0) {
      if (idx < NBOX) {
        visited[idx] = 1;
        s_kept = (validf[idx] && !suppressed[idx]) ? 1 : 0;
      } else s_kept = 0;
    }
    __syncthreads();
    if (s_kept) {
      const float* br = out + (size_t)idx * 84;
      const float ix1 = br[0], iy1 = br[1], ix2 = br[2], iy2 = br[3];
      const float iarea = (ix2 - ix1) * (iy2 - iy1);
      const int   icls  = clsi[idx];
      for (int j = tid; j < NBOX; j += 1024) {
        if (j == idx || !validf[j] || clsi[j] != icls) continue;
        const float* bj = out + (size_t)j * 84;
        const float jx1 = bj[0], jy1 = bj[1], jx2 = bj[2], jy2 = bj[3];
        const float jarea = (jx2 - jx1) * (jy2 - jy1);
        const float xx1 = fmaxf(ix1, jx1), yy1 = fmaxf(iy1, jy1);
        const float xx2 = fminf(ix2, jx2), yy2 = fminf(iy2, jy2);
        const float w = fmaxf(1e-28f, xx2 - xx1);
        const float h = fmaxf(1e-28f, yy2 - yy1);
        const float inter = w * h;
        const float iou = inter / (iarea + jarea - inter + 1e-14f);
        if (iou > 0.6f) suppressed[j] = 1;
      }
    }
    __syncthreads();
  }
  for (int j = tid; j < NBOX; j += 1024)
    keep[j] = (validf[j] && !suppressed[j]) ? 1.0f : 0.0f;
}

// ---------------------------------------------------------------------------
extern "C" void kernel_launch(void* const* d_in, const int* in_sizes, int n_in,
                              void* d_out, int out_size, void* d_ws, size_t ws_size,
                              hipStream_t stream) {
  (void)in_sizes; (void)n_in; (void)out_size; (void)ws_size;
  const float* feat   = (const float*)d_in[0];
  const float* w_conv[6] = { (const float*)d_in[1], (const float*)d_in[3],
                             (const float*)d_in[5], (const float*)d_in[7],
                             (const float*)d_in[9], (const float*)d_in[11] };
  const float* b_conv[6] = { (const float*)d_in[2], (const float*)d_in[4],
                             (const float*)d_in[6], (const float*)d_in[8],
                             (const float*)d_in[10], (const float*)d_in[12] };
  const float* w_obj  = (const float*)d_in[13]; const float* b_obj  = (const float*)d_in[14];
  const float* w_clsp = (const float*)d_in[15]; const float* b_clsp = (const float*)d_in[16];
  const float* w_regp = (const float*)d_in[17]; const float* b_regp = (const float*)d_in[18];

  float* out = (float*)d_out;
  float* ws  = (float*)d_ws;

  // ---- workspace layout (float units; all 16B-aligned chunks) ----
  float* wbf_f     = ws;                         // WELEM halfs = WELEM/2 floats
  float* partials  = wbf_f + WELEM / 2;          // 3*ACT f32
  float* featbf_f  = partials + 3 * (size_t)ACT; // ACT halfs
  float* bfA_f     = featbf_f + ACT / 2;
  float* bfB_f     = bfA_f + ACT / 2;
  float* zeroc_f   = bfB_f + ACT / 2;            // 512 halfs (256 floats)
  float* f32tmp    = zeroc_f + 256;
  float* cls_f     = f32tmp + ACT;
  float* reg_f     = cls_f + ACT;
  float* obj_raw   = reg_f + ACT;
  float* cls_raw   = obj_raw + (size_t)KA * HW;
  float* reg_raw   = cls_raw + (size_t)KA * NCLS * HW;
  float* sc        = reg_raw + (size_t)KA * 4 * HW;
  int*   clsi      = (int*)(sc + NBOX);
  int*   validf    = clsi + NBOX;
  int*   suppr     = validf + NBOX;
  int*   visited   = suppr + NBOX;

  unsigned short* wbf    = (unsigned short*)wbf_f;
  unsigned short* featbf = (unsigned short*)featbf_f;
  unsigned short* bfA    = (unsigned short*)bfA_f;
  unsigned short* bfB    = (unsigned short*)bfB_f;
  unsigned short* zeroc  = (unsigned short*)zeroc_f;

  featprep<<<(ACT + 255) / 256, 256, 0, stream>>>(feat, featbf, zeroc);

  // conv chain: cls1, cls2 | reg1, reg2, reg3, reg4
  const unsigned short* cin[6]  = { featbf, bfA, featbf, bfA, bfB, bfA };
  unsigned short*       cbf[6]  = { bfA, bfB, bfA, bfB, bfA, bfB };
  float*                cf32[6] = { f32tmp, cls_f, f32tmp, f32tmp, f32tmp, reg_f };

  const dim3 cgrid(8, 4, 3);
  for (int s = 0; s < 6; ++s) {
    wprep<<<(WELEM + 255) / 256, 256, 0, stream>>>(w_conv[s], wbf);
    conv3x3_frag<<<cgrid, 256, 0, stream>>>(cin[s], wbf, zeroc, partials);
    combine_act<<<(ACT + 255) / 256, 256, 0, stream>>>(partials, b_conv[s], cf32[s], cbf[s]);
  }

  conv1x1<<<(KA * HW + 255) / 256, 256, 0, stream>>>(reg_f, w_obj, b_obj, obj_raw, KA);
  conv1x1<<<(KA * NCLS * HW + 255) / 256, 256, 0, stream>>>(cls_f, w_clsp, b_clsp, cls_raw, KA * NCLS);
  conv1x1<<<(KA * 4 * HW + 255) / 256, 256, 0, stream>>>(reg_f, w_regp, b_regp, reg_raw, KA * 4);

  decode_kernel<<<(NBOX + 127) / 128, 128, 0, stream>>>(
      obj_raw, cls_raw, reg_raw, out, sc, clsi, validf, suppr, visited);

  nms_kernel<<<1, 1024, 0, stream>>>(out, sc, clsi, validf, suppr, visited,
                                     out + (size_t)NBOX * 84);
}
